// peepLSTM_13795434955473
// MI455X (gfx1250) — matrix operand order repaired and checked by reading, not compiled, not run on hardware
//
#include <hip/hip_runtime.h>
#include <hip/hip_bf16.h>

typedef __attribute__((ext_vector_type(16))) __bf16    v16bf;
typedef __attribute__((ext_vector_type(8)))  float     v8f;
typedef __attribute__((ext_vector_type(4)))  unsigned  v4u;

#define Hdim 768
#define Bdim 512
#define Edim 6
#define Cdim 10
#define Vdim 3
#define Sdim 256

union Frag { v4u u[2]; v16bf v; };

__device__ __forceinline__ unsigned short f2bf(float f) {
    unsigned u = __float_as_uint(f);
    unsigned r = u + 0x7FFFu + ((u >> 16) & 1u);   // round-to-nearest-even
    return (unsigned short)(r >> 16);
}
__device__ __forceinline__ float sigmoidf(float x) {
    return 1.0f / (1.0f + __expf(-x));
}

// ---- prep: f32 recurrent weights -> bf16, row-major, 3 gates stacked ----
__global__ void pack_wh_kernel(const float* __restrict__ wf,
                               const float* __restrict__ wi,
                               const float* __restrict__ wo,
                               unsigned short* __restrict__ dst) {
    int idx = blockIdx.x * 256 + threadIdx.x;
    const int n = Hdim * Hdim;
    if (idx >= 3 * n) return;
    const float* src = (idx < n) ? wf : ((idx < 2 * n) ? wi : wo);
    int off = (idx < n) ? idx : ((idx < 2 * n) ? idx - n : idx - 2 * n);
    dst[idx] = f2bf(src[off]);
}

// ---- prep: gate-x lookup tables tab[g][v][m] = b_g[m] + W_gx[m,:].emb[v,:] ----
__global__ void pack_tab_kernel(const float* __restrict__ emb,
                                const float* __restrict__ Wfx, const float* __restrict__ bf,
                                const float* __restrict__ Wix, const float* __restrict__ bi,
                                const float* __restrict__ Wox, const float* __restrict__ bo,
                                const float* __restrict__ Wcx, const float* __restrict__ bc,
                                float* __restrict__ tab) {
    int idx = blockIdx.x * 256 + threadIdx.x;
    if (idx >= 4 * Vdim * Hdim) return;
    int m = idx % Hdim;
    int v = (idx / Hdim) % Vdim;
    int g = idx / (Vdim * Hdim);
    const float* W = (g == 0) ? Wfx : (g == 1) ? Wix : (g == 2) ? Wox : Wcx;
    const float* b = (g == 0) ? bf  : (g == 1) ? bi  : (g == 2) ? bo  : bc;
    float s = b[m];
    for (int e = 0; e < Edim; ++e) s += W[m * Edim + e] * emb[v * Edim + e];
    tab[idx] = s;
}

// ---- prep: c_init -> f32 state copy + bf16 B-fragment-packed copy ----
__global__ void pack_c_kernel(const float* __restrict__ cin,
                              float* __restrict__ cf,
                              unsigned short* __restrict__ cpk) {
    int idx = blockIdx.x * 256 + threadIdx.x;
    if (idx >= Hdim * Bdim) return;
    int m = idx / Bdim, b = idx % Bdim;
    float c = cin[idx];
    cf[idx] = c;
    int kc = m >> 5, kr = m & 31, nt = b >> 4, nn = b & 15;
    cpk[(kc * (Bdim / 16) + nt) * 512 + (nn + (kr & 16)) * 16 + (kr & 15)] = f2bf(c);
}

// ---- fragment loaders (indices in v4u = 8 halves units) ----
__device__ __forceinline__ void load_frags(const v4u* __restrict__ A,
                                           const v4u* __restrict__ Bp,
                                           int aBase, int bBase, int kc,
                                           Frag (&af)[3], Frag (&bfr)[2]) {
    const int gstr = (Hdim * Hdim) / 8;          // gate stride in v4u
    int bIdx = bBase + kc * ((Bdim / 16) * 64);  // next K-chunk of packed B
    bfr[0].u[0] = Bp[bIdx];      bfr[0].u[1] = Bp[bIdx + 1];
    bfr[1].u[0] = Bp[bIdx + 64]; bfr[1].u[1] = Bp[bIdx + 65];
    int aIdx = aBase + kc * 4;                   // 32 halves = 4 v4u per K-chunk
    af[0].u[0] = A[aIdx];             af[0].u[1] = A[aIdx + 1];
    af[1].u[0] = A[aIdx + gstr];      af[1].u[1] = A[aIdx + gstr + 1];
    af[2].u[0] = A[aIdx + 2 * gstr];  af[2].u[1] = A[aIdx + 2 * gstr + 1];
}

__device__ __forceinline__ void do_mma(const Frag (&af)[3], const Frag (&bfr)[2],
                                       v8f (&acc)[3][2]) {
    #pragma unroll
    for (int g = 0; g < 3; ++g)
        #pragma unroll
        for (int j = 0; j < 2; ++j)
            acc[g][j] = __builtin_amdgcn_wmma_f32_16x16x32_bf16(
                false, af[g].v, false, bfr[j].v, (short)0, acc[g][j], false, false);
}

// ---- one LSTM step: G = W_{f,i,o}h @ c_prev (bf16 WMMA, f32 acc), fused gates ----
__global__ __launch_bounds__(256) void lstm_step_kernel(
    const unsigned short* __restrict__ Wh,    // [3][H][H] bf16 row-major
    const unsigned short* __restrict__ cpk,   // packed bf16 c_prev (B-fragment layout)
    const float* __restrict__ cf,             // f32 c_prev [H][B]
    const float* __restrict__ tab,            // [4][V][H]
    const long long* __restrict__ x,          // [B][S] int64 tokens
    int t,
    float* __restrict__ cf_out,
    unsigned short* __restrict__ cpk_out,
    float* __restrict__ h_out)                // nullptr except last step
{
    const int lane = threadIdx.x & 31;
    const int w    = threadIdx.x >> 5;
    const int mrow = blockIdx.x * 64 + (w & 3) * 16;   // 12 blocks * 64 = 768
    const int ncol = blockIdx.y * 64 + (w >> 2) * 32;  // 8 blocks  * 64 = 512

    v8f acc[3][2];
    #pragma unroll
    for (int g = 0; g < 3; ++g)
        #pragma unroll
        for (int j = 0; j < 2; ++j)
            #pragma unroll
            for (int r = 0; r < 8; ++r) acc[g][j][r] = 0.0f;

    const v4u* A  = (const v4u*)Wh;
    const v4u* Bp = (const v4u*)cpk;
    const int arow  = mrow + (lane & 15);
    const int asel  = (lane & 16) ? 16 : 0;            // hi half-wave reads K+16..31
    const int aBase = (arow * Hdim + asel) >> 3;       // v4u index at kc=0, gate 0
    const int bBase = (ncol >> 4) * 64 + lane * 2;     // v4u index at kc=0, tile j=0

    // Double-buffered K loop: overlap next chunk's loads with current WMMAs.
    Frag a0[3], b0[2], a1[3], b1[2];
    load_frags(A, Bp, aBase, bBase, 0, a0, b0);
    const int KC = Hdim / 32;                          // 24 chunks
    for (int kc = 0; kc < KC; kc += 2) {
        load_frags(A, Bp, aBase, bBase, kc + 1, a1, b1);
        do_mma(a0, b0, acc);
        if (kc + 2 < KC)
            load_frags(A, Bp, aBase, bBase, kc + 2, a0, b0);
        do_mma(a1, b1, acc);
    }

    // Epilogue: gates + peephole update. C/D layout: VGPR r -> M = r + 8*(lane>=16), N = lane&15.
    const int mB = (lane & 16) ? 8 : 0;
    for (int j = 0; j < 2; ++j) {
        const int b   = ncol + j * 16 + (lane & 15);
        const int tok = (int)x[b * Sdim + t];
        const float* tf = tab + (0 * Vdim + tok) * Hdim;
        const float* ti = tab + (1 * Vdim + tok) * Hdim;
        const float* to = tab + (2 * Vdim + tok) * Hdim;
        const float* tc = tab + (3 * Vdim + tok) * Hdim;
        #pragma unroll
        for (int r = 0; r < 8; ++r) {
            int m   = mrow + mB + r;
            float f = sigmoidf(acc[0][j][r] + tf[m]);
            float i = sigmoidf(acc[1][j][r] + ti[m]);
            float o = sigmoidf(acc[2][j][r] + to[m]);
            float cc = sigmoidf(tc[m]);                 // candidate gate: lookup only
            float cp = cf[m * Bdim + b];
            float cn = cc * i + cp * f;
            cf_out[m * Bdim + b] = cn;
            int kc2 = m >> 5, kr = m & 31, nt = b >> 4, nn = b & 15;
            cpk_out[(kc2 * (Bdim / 16) + nt) * 512 + (nn + (kr & 16)) * 16 + (kr & 15)] = f2bf(cn);
            if (h_out) h_out[m * Bdim + b] = tanhf(cn) * o;
        }
    }
}

// ---- final: p = W_ph @ h + b_p ; out = log_softmax(p^T) ----
__global__ void final_proj_kernel(const float* __restrict__ Wph,
                                  const float* __restrict__ bp,
                                  const float* __restrict__ h,
                                  float* __restrict__ out) {
    int b = blockIdx.x * 256 + threadIdx.x;
    if (b >= Bdim) return;
    float p[Cdim];
    float mx = -1e30f;
    for (int c = 0; c < Cdim; ++c) {
        float s = bp[c];
        for (int m = 0; m < Hdim; ++m) s += Wph[c * Hdim + m] * h[m * Bdim + b];
        p[c] = s;
        mx = fmaxf(mx, s);
    }
    float sum = 0.0f;
    for (int c = 0; c < Cdim; ++c) sum += __expf(p[c] - mx);
    float lse = mx + __logf(sum);
    for (int c = 0; c < Cdim; ++c) out[b * Cdim + c] = p[c] - lse;
}

extern "C" void kernel_launch(void* const* d_in, const int* in_sizes, int n_in,
                              void* d_out, int out_size, void* d_ws, size_t ws_size,
                              hipStream_t stream) {
    (void)in_sizes; (void)n_in; (void)out_size; (void)ws_size;
    const long long* x   = (const long long*)d_in[0];
    const float* emb     = (const float*)d_in[1];
    const float* W_fx    = (const float*)d_in[2];
    const float* W_fh    = (const float*)d_in[3];
    const float* b_f     = (const float*)d_in[4];
    const float* W_ix    = (const float*)d_in[5];
    const float* W_ih    = (const float*)d_in[6];
    const float* b_i     = (const float*)d_in[7];
    const float* W_ox    = (const float*)d_in[8];
    const float* W_oh    = (const float*)d_in[9];
    const float* b_o     = (const float*)d_in[10];
    const float* W_cx    = (const float*)d_in[11];
    const float* b_c     = (const float*)d_in[12];
    const float* W_ph    = (const float*)d_in[13];
    const float* b_p     = (const float*)d_in[14];
    const float* c_init  = (const float*)d_in[15];

    char* ws = (char*)d_ws;
    size_t off = 0;
    auto alloc = [&](size_t bytes) -> void* {
        void* p = ws + off;
        off = (off + bytes + 255) & ~(size_t)255;
        return p;
    };
    unsigned short* Wh   = (unsigned short*)alloc((size_t)3 * Hdim * Hdim * 2);
    float*          tab  = (float*)alloc((size_t)4 * Vdim * Hdim * 4);
    float*          cf0  = (float*)alloc((size_t)Hdim * Bdim * 4);
    float*          cf1  = (float*)alloc((size_t)Hdim * Bdim * 4);
    unsigned short* cpk0 = (unsigned short*)alloc((size_t)Hdim * Bdim * 2);
    unsigned short* cpk1 = (unsigned short*)alloc((size_t)Hdim * Bdim * 2);
    float*          hbuf = (float*)alloc((size_t)Hdim * Bdim * 4);

    pack_wh_kernel<<<(3 * Hdim * Hdim + 255) / 256, 256, 0, stream>>>(W_fh, W_ih, W_oh, Wh);
    pack_tab_kernel<<<(4 * Vdim * Hdim + 255) / 256, 256, 0, stream>>>(
        emb, W_fx, b_f, W_ix, b_i, W_ox, b_o, W_cx, b_c, tab);
    pack_c_kernel<<<(Hdim * Bdim + 255) / 256, 256, 0, stream>>>(c_init, cf0, cpk0);

    dim3 grid(Hdim / 64, Bdim / 64, 1);
    for (int t = 0; t < Sdim - 1; ++t) {
        int s = t & 1;
        lstm_step_kernel<<<grid, 256, 0, stream>>>(
            Wh,
            s ? cpk1 : cpk0, s ? cf1 : cf0,
            tab, x, t,
            s ? cf0 : cf1, s ? cpk0 : cpk1,
            (t == Sdim - 2) ? hbuf : nullptr);
    }
    final_proj_kernel<<<(Bdim + 255) / 256, 256, 0, stream>>>(W_ph, b_p, hbuf, (float*)d_out);
}
